// SNN_50672024158347
// MI455X (gfx1250) — matrix operand order repaired and checked by reading, not compiled, not run on hardware
//
#include <hip/hip_runtime.h>

// ---------------------------------------------------------------------------
// Fused SNN layer for MI455X (gfx1250):
//   gs[b,t,m] = sum_k input[b,t,k] * W[m,k] + bias[m]     (bf16 WMMA, f32 acc)
//   LIF scan over t:  v = v + dt*((0 - v) + g);  z = (v > 1);  v -= z
//
// B=32, T=128, F=K=2048.  One workgroup = one (batch, 256-wide F tile):
// a 128(T) x 256(F) x 2048(K) GEMM tile kept entirely on-chip, then the
// sequential scan is run out of LDS and spikes stored coalesced.
// ---------------------------------------------------------------------------

#define B_DIM 32
#define T_DIM 128
#define F_DIM 2048
#define K_DIM 2048
#define FTILE 256      // F columns per workgroup
#define KC    64       // K chunk staged in LDS per iteration
#define GS_STRIDE 260  // 256 + 4 pad floats -> conflict-free C scatter & scan

typedef __attribute__((ext_vector_type(16))) __bf16 v16bf;
typedef __attribute__((ext_vector_type(8)))  float  v8f;

union FragAB {
    uint4 q[2];   // two 16B LDS reads
    v16bf v;      // WMMA operand
};
static_assert(sizeof(FragAB) == 32, "frag size");

// f32 -> bf16 round-to-nearest-even, packed pair into one dword
__device__ __forceinline__ unsigned int pack2_bf16(float a, float b) {
    union { float f; unsigned int u; } x, y;
    x.f = a; y.f = b;
    unsigned int ra = x.u + 0x7FFFu + ((x.u >> 16) & 1u);
    unsigned int rb = y.u + 0x7FFFu + ((y.u >> 16) & 1u);
    return (ra >> 16) | (rb & 0xFFFF0000u);
}

extern "C" __global__ __launch_bounds__(256)
void snn_lif_fused_kernel(const float* __restrict__ inp,   // [B,T,K]
                          const float* __restrict__ Wm,    // [F,K]
                          const float* __restrict__ bias,  // [F]
                          float* __restrict__ out)         // [B,T,F]
{
    extern __shared__ unsigned char smem[];
    // Staging layout (phase 1):  A_lds [128][KC] bf16 (16KB), W_lds [256][KC] bf16 (32KB)
    // gs layout     (phase 2):  [128][GS_STRIDE] f32 (130KB), aliases staging
    unsigned short* A_lds = (unsigned short*)smem;
    unsigned short* W_lds = (unsigned short*)(smem + T_DIM * KC * sizeof(unsigned short));
    float*          gs    = (float*)smem;

    const int tid  = threadIdx.x;
    const int lane = tid & 31;
    const int wid  = tid >> 5;          // 8 waves
    const int m_w  = wid >> 2;          // 0..1 : 64 rows (T) per wave
    const int n_w  = wid & 3;           // 0..3 : 64 cols (F) per wave
    const int laneM = lane & 15;
    const int lo_half = (lane < 16);

    const int ft = blockIdx.x;          // F tile index (0..7)
    const int bb = blockIdx.y;          // batch (0..31)
    const int f0 = ft * FTILE;

    const float* Ab = inp + (size_t)bb * T_DIM * K_DIM;  // A[t][k]
    const float* Wb = Wm  + (size_t)f0 * K_DIM;          // B rows: W[f0+n][k]

    v8f acc[4][4];
    const v8f vzero = {0.f, 0.f, 0.f, 0.f, 0.f, 0.f, 0.f, 0.f};
#pragma unroll
    for (int mt = 0; mt < 4; ++mt)
#pragma unroll
        for (int nt = 0; nt < 4; ++nt)
            acc[mt][nt] = vzero;

    const int abase = lo_half ? 0 : 8;   // ISA 16-bit A 16x32 lane K-base
    const int bbase = lo_half ? 0 : 8;   // ISA 16-bit B 32x16 lane K-base
    const int mrow0 = m_w * 64 + laneM;
    const int ncol0 = n_w * 64 + laneM;

    // ---------------- GEMM: stream K in KC-wide chunks --------------------
    for (int k0 = 0; k0 < K_DIM; k0 += KC) {
        __syncthreads();  // previous chunk's LDS fully consumed

        // Stage A: 128 rows x 16 float4 -> bf16 LDS  (8 iters/thread)
        const float* Ac = Ab + k0;
        for (int i = tid; i < T_DIM * (KC / 4); i += 256) {
            const int t = i >> 4, c = i & 15;
            float4 v = *(const float4*)(Ac + (size_t)t * K_DIM + c * 4);
            unsigned int p0 = pack2_bf16(v.x, v.y);
            unsigned int p1 = pack2_bf16(v.z, v.w);
            *(uint2*)(&A_lds[t * KC + c * 4]) = make_uint2(p0, p1);
        }
        // Stage W: 256 rows x 16 float4 -> bf16 LDS  (16 iters/thread)
        const float* Wc = Wb + k0;
        for (int i = tid; i < FTILE * (KC / 4); i += 256) {
            const int n = i >> 4, c = i & 15;
            float4 v = *(const float4*)(Wc + (size_t)n * K_DIM + c * 4);
            unsigned int p0 = pack2_bf16(v.x, v.y);
            unsigned int p1 = pack2_bf16(v.z, v.w);
            *(uint2*)(&W_lds[n * KC + c * 4]) = make_uint2(p0, p1);
        }
        // Prefetch next chunk's rows into cache (global_prefetch_b8)
        if (k0 + KC < K_DIM) {
            const float* An = Ac + KC;
            const float* Wn = Wc + KC;
            for (int i = tid; i < T_DIM; i += 256) __builtin_prefetch(An + (size_t)i * K_DIM, 0, 0);
            for (int i = tid; i < FTILE; i += 256) __builtin_prefetch(Wn + (size_t)i * K_DIM, 0, 0);
        }
        __syncthreads();

        // Two WMMA K-steps of 32 per chunk
#pragma unroll
        for (int ks = 0; ks < 2; ++ks) {
            const int k32 = ks * 32;
            FragAB afr[4], bfr[4];
#pragma unroll
            for (int mt = 0; mt < 4; ++mt) {
                const unsigned short* p = A_lds + (mrow0 + mt * 16) * KC + k32 + abase;
                afr[mt].q[0] = *(const uint4*)(p);        // K = base+0..7
                afr[mt].q[1] = *(const uint4*)(p + 16);   // K = base+16..23
            }
#pragma unroll
            for (int nt = 0; nt < 4; ++nt) {
                const unsigned short* p = W_lds + (ncol0 + nt * 16) * KC + k32 + bbase;
                bfr[nt].q[0] = *(const uint4*)(p);        // K = base+0..7
                bfr[nt].q[1] = *(const uint4*)(p + 16);   // K = base+16..23
            }
#pragma unroll
            for (int mt = 0; mt < 4; ++mt)
#pragma unroll
                for (int nt = 0; nt < 4; ++nt)
                    acc[mt][nt] = __builtin_amdgcn_wmma_f32_16x16x32_bf16(
                        false, afr[mt].v, false, bfr[nt].v,
                        (short)0, acc[mt][nt], false, false);
        }
    }

    // ---------------- Scatter C fragments into gs LDS ---------------------
    __syncthreads();  // done reading staging LDS; safe to alias as gs
    {
        const int thalf = lo_half ? 0 : 8;  // ISA C/D layout: VGPR v -> M = v (+8)
#pragma unroll
        for (int mt = 0; mt < 4; ++mt) {
#pragma unroll
            for (int nt = 0; nt < 4; ++nt) {
                const int f = n_w * 64 + nt * 16 + laneM;
                const int t0 = m_w * 64 + mt * 16 + thalf;
#pragma unroll
                for (int v = 0; v < 8; ++v)
                    gs[(t0 + v) * GS_STRIDE + f] = acc[mt][nt][v];
            }
        }
    }
    __syncthreads();

    // ---------------- LIF scan over T (sequential) ------------------------
    {
        const float dt  = 1.0f / 100.0f;   // 1/TAU_MEM
        const float bsv = bias[f0 + tid];
        float v = 0.0f;
        float* op = out + (size_t)bb * T_DIM * F_DIM + f0 + tid;
        for (int t = 0; t < T_DIM; ++t) {
            const float g = gs[t * GS_STRIDE + tid] + bsv;
            v = v + dt * ((0.0f - v) + g);           // Euler LIF update
            const float z = (v > 1.0f) ? 1.0f : 0.0f;
            v -= z;                                  // subtractive reset
            op[(size_t)t * F_DIM] = z;
        }
    }
}

extern "C" void kernel_launch(void* const* d_in, const int* in_sizes, int n_in,
                              void* d_out, int out_size, void* d_ws, size_t ws_size,
                              hipStream_t stream) {
    (void)in_sizes; (void)n_in; (void)out_size; (void)d_ws; (void)ws_size;
    const float* inp  = (const float*)d_in[0];
    const float* Wm   = (const float*)d_in[1];
    const float* bias = (const float*)d_in[2];
    float* out = (float*)d_out;

    dim3 grid(F_DIM / FTILE, B_DIM);   // (8, 32)
    dim3 block(256);
    size_t lds = (size_t)T_DIM * GS_STRIDE * sizeof(float);  // 130 KB (aliases 48 KB staging)
    snn_lif_fused_kernel<<<grid, block, lds, stream>>>(inp, Wm, bias, out);
}
